// HNHNLayer_68667937128453
// MI455X (gfx1250) — compile-verified, matrix operand order checked by reading, not run on hardware
//
#include <hip/hip_runtime.h>
#include <hip/hip_bf16.h>

#define N_NODES 16384
#define N_EDGES 8192
#define CH 256

typedef __bf16 bf16_t;
typedef bf16_t v16bf __attribute__((ext_vector_type(16)));
typedef bf16_t bf16x4 __attribute__((ext_vector_type(4)));
typedef float  v8f   __attribute__((ext_vector_type(8)));
typedef unsigned int u32x4 __attribute__((ext_vector_type(4)));
typedef int v4i __attribute__((ext_vector_type(4)));

#if __has_builtin(__builtin_amdgcn_global_load_async_to_lds_b128)
#define HAVE_ASYNC_LDS 1
#else
#define HAVE_ASYNC_LDS 0
#endif

// 16-byte global -> LDS async copy (ASYNCcnt-tracked), if available.
// Builtin signature (from compiler diagnostic): (v4i AS1*, v4i AS3*, imm, imm)
__device__ static inline void async_cp16(const void* g, void* l) {
#if HAVE_ASYNC_LDS
    __builtin_amdgcn_global_load_async_to_lds_b128(
        (__attribute__((address_space(1))) v4i*)(g),
        (__attribute__((address_space(3))) v4i*)(l), 0, 0);
#endif
}

__device__ static inline void wait_async() {
#if HAVE_ASYNC_LDS
#if __has_builtin(__builtin_amdgcn_s_wait_asynccnt)
    __builtin_amdgcn_s_wait_asynccnt(0);
#else
    asm volatile("s_wait_asynccnt 0" ::: "memory");
#endif
#endif
}

// ---------------------------------------------------------------------------
// Normalization pass kernels (streaming over B1, memory-bound)
// ---------------------------------------------------------------------------

__global__ __launch_bounds__(256) void k_zero(float* p, int n) {
    int i = blockIdx.x * 256 + threadIdx.x;
    if (i < n) p[i] = 0.0f;
}

// Pass 1 over B1: column sums (edge degree) + row sums (node degree).
__global__ __launch_bounds__(256)
void k_degrees(const float* __restrict__ B1, float* __restrict__ colsum,
               float* __restrict__ rowsum) {
    const int lane   = threadIdx.x & 31;
    const int wv     = threadIdx.x >> 5;
    const int colBlk = blockIdx.x & 31;   // 32 blocks of 256 columns
    const int slab   = blockIdx.x >> 5;   // 8 slabs of 2048 rows
    const int e      = colBlk * 256 + wv * 32 + lane;
    const int nBeg   = slab * 2048, nEnd = nBeg + 2048;
    float cacc = 0.0f;
    for (int n = nBeg; n < nEnd; ++n) {
        float v = B1[(size_t)n * N_EDGES + e];
        cacc += v;
        float r = v;
        for (int off = 16; off; off >>= 1) r += __shfl_xor(r, off);
        if (lane == 0) atomicAdd(&rowsum[n], r);
    }
    atomicAdd(&colsum[e], cacc);
}

// colsum -> edge_card = s^-1.5, rowsum -> node_card = s^-0.5 (in place)
__global__ __launch_bounds__(256) void k_card(float* edge_card, float* node_card) {
    int i = blockIdx.x * 256 + threadIdx.x;
    if (i < N_EDGES) {
        float s = edge_card[i];
        float rc = rsqrtf(s);
        edge_card[i] = rc * rc * rc;          // s^-1.5
    }
    if (i < N_NODES) node_card[i] = rsqrtf(node_card[i]);
}

// Pass 2 over B1: d1acc[e] = sum_n B1[n,e]*node_card[n],
//                 d0acc[n] = sum_e B1[n,e]*edge_card[e]
__global__ __launch_bounds__(256)
void k_dacc(const float* __restrict__ B1, const float* __restrict__ edge_card,
            const float* __restrict__ node_card,
            float* __restrict__ d0acc, float* __restrict__ d1acc) {
    const int lane   = threadIdx.x & 31;
    const int wv     = threadIdx.x >> 5;
    const int colBlk = blockIdx.x & 31;
    const int slab   = blockIdx.x >> 5;
    const int e      = colBlk * 256 + wv * 32 + lane;
    const int nBeg   = slab * 2048, nEnd = nBeg + 2048;
    const float ec = edge_card[e];
    float dacc1 = 0.0f;
    for (int n = nBeg; n < nEnd; ++n) {
        float v  = B1[(size_t)n * N_EDGES + e];
        dacc1 += v * node_card[n];
        float r = v * ec;
        for (int off = 16; off; off >>= 1) r += __shfl_xor(r, off);
        if (lane == 0) atomicAdd(&d0acc[n], r);
    }
    atomicAdd(&d1acc[e], dacc1);
}

__global__ __launch_bounds__(256) void k_invert(float* d0, float* d1) {
    int i = blockIdx.x * 256 + threadIdx.x;
    if (i < N_EDGES) d1[i] = 1.0f / d1[i];
    if (i < N_NODES) d0[i] = 1.0f / d0[i];
}

// ---------------------------------------------------------------------------
// Precision conversion helpers
// ---------------------------------------------------------------------------

__global__ __launch_bounds__(256) void k_convW(const float* __restrict__ W,
                                               bf16_t* __restrict__ Wt) {
    int i = blockIdx.x * 256 + threadIdx.x;   // 65536 threads
    int c = i & 255, k = i >> 8;
    Wt[(size_t)c * 256 + k] = (bf16_t)W[(size_t)k * 256 + c];
}

__global__ __launch_bounds__(256) void k_convX(const float* __restrict__ X,
                                               bf16_t* __restrict__ Xb, int n) {
    int i = blockIdx.x * 256 + threadIdx.x;
    if (i < n) Xb[i] = (bf16_t)X[i];
}

// ---------------------------------------------------------------------------
// Small GEMM:  Ct[c][m] = scale[m] * sum_k Ab[m,k] * Wt[c,k]
// ---------------------------------------------------------------------------
__global__ __launch_bounds__(256)
void gemm_xw(const bf16_t* __restrict__ Ab,   // [M][256] bf16 row-major
             const bf16_t* __restrict__ Wt,   // [256][256] bf16 (c-major)
             const float*  __restrict__ scale,// [M]
             bf16_t* __restrict__ Ct,         // [256][M]
             int M) {
    constexpr int LDA = 72;                   // padded bf16 stride
    __shared__ bf16_t As[64 * LDA];
    __shared__ bf16_t Bs[256 * LDA];

    const int tid  = threadIdx.x;
    const int lane = tid & 31;
    const int wave = tid >> 5;
    const int m0   = blockIdx.x * 64;

    v8f acc[8];
    const v8f vzero = {0.f, 0.f, 0.f, 0.f, 0.f, 0.f, 0.f, 0.f};
    for (int j = 0; j < 8; ++j) acc[j] = vzero;

    const int mt  = (wave & 3) * 16;
    const int cb  = (wave >> 2) * 128;
    const int lo  = lane & 15;
    const int kh  = (lane >> 4) * 8;
    const int tx  = tid & 7, ty = tid >> 3;

    for (int k0 = 0; k0 < 256; k0 += 64) {
        __syncthreads();
        for (int i = 0; i < 8; ++i) {
            int c = ty + i * 32;
            u32x4 d = *(const u32x4*)(Wt + (size_t)c * 256 + k0 + tx * 8);
            *(u32x4*)(&Bs[c * LDA + tx * 8]) = d;
        }
        for (int i = 0; i < 2; ++i) {
            int ml = ty + i * 32;
            u32x4 d = *(const u32x4*)(Ab + (size_t)(m0 + ml) * 256 + k0 + tx * 8);
            *(u32x4*)(&As[ml * LDA + tx * 8]) = d;
        }
        __syncthreads();

        const int arow = mt + lo;
        for (int kk = 0; kk < 64; kk += 32) {
            union { v16bf v; u32x4 u[2]; } a;
            a.u[0] = *(const u32x4*)(&As[arow * LDA + kk + kh]);
            a.u[1] = *(const u32x4*)(&As[arow * LDA + kk + 16 + kh]);
            for (int j = 0; j < 8; ++j) {
                int c = cb + j * 16 + lo;
                union { v16bf v; u32x4 u[2]; } b;
                b.u[0] = *(const u32x4*)(&Bs[c * LDA + kk + kh]);
                b.u[1] = *(const u32x4*)(&Bs[c * LDA + kk + 16 + kh]);
                acc[j] = __builtin_amdgcn_wmma_f32_16x16x32_bf16(
                    false, a.v, false, b.v, (short)0, acc[j], false, false);
            }
        }
    }

    const int hi = (lane >> 4) * 8;
    float sc[8];
    for (int r = 0; r < 8; ++r) sc[r] = scale[m0 + mt + r + hi];
    for (int j = 0; j < 8; ++j) {
        int c = cb + j * 16 + lo;
        for (int r = 0; r < 8; ++r) {
            int m = m0 + mt + r + hi;
            Ct[(size_t)c * M + m] = (bf16_t)(acc[j][r] * sc[r]);
        }
    }
}

// ---------------------------------------------------------------------------
// Big incidence GEMM (HBM-bound workhorse), double-buffered LDS pipeline.
//   TA=true : C[e,c] = d1inv[e]*(sum_n B1[n,e]*Bt[c,n]) + bias[c]
//   TA=false: C[n,c] = d0inv[n]*(sum_e B1[n,e]*Bt[c,e]) + bias[c]
// ---------------------------------------------------------------------------
template <bool TA>
__global__ __launch_bounds__(256)
void big_gemm(const float*  __restrict__ B1,    // [N_NODES][N_EDGES] fp32
              const bf16_t* __restrict__ Bt,    // [CH][Kdim] bf16
              const float*  __restrict__ scale, // [M]
              const float*  __restrict__ bias,  // [CH]
              float*  __restrict__ outRelu,     // [M][CH]
              bf16_t* __restrict__ outBf) {     // [M][CH] (TA only)
    constexpr int Kdim   = TA ? N_NODES : N_EDGES;
    constexpr int NSTAGE = Kdim / 64;
    constexpr int LDA    = 72;
    __shared__ bf16_t As[2][64 * LDA];
    __shared__ bf16_t Bs[2][256 * LDA];

    const int tid  = threadIdx.x;
    const int lane = tid & 31;
    const int wave = tid >> 5;
    const int m0   = blockIdx.x * 64;

    v8f acc[8];
    const v8f vzero = {0.f, 0.f, 0.f, 0.f, 0.f, 0.f, 0.f, 0.f};
    for (int j = 0; j < 8; ++j) acc[j] = vzero;

    const int mt = (wave & 3) * 16;
    const int cb = (wave >> 2) * 128;
    const int lo = lane & 15;
    const int kh = (lane >> 4) * 8;
    const int tx = tid & 7, ty = tid >> 3;

    // ---- stage loader: global -> LDS buffer `buf` for K-offset k0 ----
    auto load_stage = [&](int k0, int buf) {
        // B tile (channel-major Yt/Zt, bf16): pure copy -> async if available
        for (int i = 0; i < 8; ++i) {
            int c = ty + i * 32;
#if HAVE_ASYNC_LDS
            async_cp16(Bt + (size_t)c * Kdim + k0 + tx * 8,
                       &Bs[buf][c * LDA + tx * 8]);
#else
            u32x4 d = *(const u32x4*)(Bt + (size_t)c * Kdim + k0 + tx * 8);
            *(u32x4*)(&Bs[buf][c * LDA + tx * 8]) = d;
#endif
        }
        // A tile from B1 (fp32 -> bf16 conversion; binary values -> exact)
        if (TA) {
            // As[e_local][n_local] = B1[k0+n][m0+e]  (transpose on store)
            int ex = (tid & 15) * 4;
            int ng = tid >> 4;
            for (int i = 0; i < 4; ++i) {
                int nl = ng + i * 16;
                const float4 f = *(const float4*)(B1 + (size_t)(k0 + nl) * N_EDGES + m0 + ex);
                As[buf][(ex + 0) * LDA + nl] = (bf16_t)f.x;
                As[buf][(ex + 1) * LDA + nl] = (bf16_t)f.y;
                As[buf][(ex + 2) * LDA + nl] = (bf16_t)f.z;
                As[buf][(ex + 3) * LDA + nl] = (bf16_t)f.w;
            }
        } else {
            // As[n_local][e_local] = B1[m0+n][k0+e]  (K contiguous)
            int kx = (tid & 15) * 4;
            int mg = tid >> 4;
            for (int i = 0; i < 4; ++i) {
                int ml = mg + i * 16;
                const float4 f = *(const float4*)(B1 + (size_t)(m0 + ml) * N_EDGES + k0 + kx);
                bf16x4 p = {(bf16_t)f.x, (bf16_t)f.y, (bf16_t)f.z, (bf16_t)f.w};
                *(bf16x4*)(&As[buf][ml * LDA + kx]) = p;
            }
        }
    };

    load_stage(0, 0);

    for (int s = 0; s < NSTAGE; ++s) {
        const int cur = s & 1;
        wait_async();           // issuing wave's async LDS writes must land
        __syncthreads();        // stage `s` visible; buffer cur^1 free

        if (s + 1 < NSTAGE) load_stage((s + 1) * 64, cur ^ 1);

        // prefetch the B1 HBM stream two stages ahead
        if (s + 2 < NSTAGE) {
            const int kp = (s + 2) * 64;
            if (TA)
                __builtin_prefetch(B1 + (size_t)(kp + (tid >> 4)) * N_EDGES + m0 + (tid & 15) * 4, 0, 1);
            else
                __builtin_prefetch(B1 + (size_t)(m0 + (tid >> 4)) * N_EDGES + kp + (tid & 15) * 4, 0, 1);
        }

        const int arow = mt + lo;
        for (int kk = 0; kk < 64; kk += 32) {
            union { v16bf v; u32x4 u[2]; } a;
            a.u[0] = *(const u32x4*)(&As[cur][arow * LDA + kk + kh]);
            a.u[1] = *(const u32x4*)(&As[cur][arow * LDA + kk + 16 + kh]);
            for (int j = 0; j < 8; ++j) {
                int c = cb + j * 16 + lo;
                union { v16bf v; u32x4 u[2]; } b;
                b.u[0] = *(const u32x4*)(&Bs[cur][c * LDA + kk + kh]);
                b.u[1] = *(const u32x4*)(&Bs[cur][c * LDA + kk + 16 + kh]);
                acc[j] = __builtin_amdgcn_wmma_f32_16x16x32_bf16(
                    false, a.v, false, b.v, (short)0, acc[j], false, false);
            }
        }
    }

    const int hi = (lane >> 4) * 8;
    float sc[8];
    for (int r = 0; r < 8; ++r) sc[r] = scale[m0 + mt + r + hi];
    for (int j = 0; j < 8; ++j) {
        int c = cb + j * 16 + lo;
        float bv = bias[c];
        for (int r = 0; r < 8; ++r) {
            int m = m0 + mt + r + hi;
            float v = sc[r] * acc[j][r] + bv;
            size_t o = (size_t)m * CH + c;
            if (TA) outBf[o] = (bf16_t)v;        // pre-relu x_1 (bf16)
            outRelu[o] = fmaxf(v, 0.0f);
        }
    }
}

// ---------------------------------------------------------------------------
// Launcher
// ---------------------------------------------------------------------------
extern "C" void kernel_launch(void* const* d_in, const int* in_sizes, int n_in,
                              void* d_out, int out_size, void* d_ws, size_t ws_size,
                              hipStream_t stream) {
    const float* x0  = (const float*)d_in[0];
    const float* B1  = (const float*)d_in[1];
    const float* W0  = (const float*)d_in[2];
    const float* W1  = (const float*)d_in[3];
    const float* b01 = (const float*)d_in[4];
    const float* b10 = (const float*)d_in[5];

    char* w = (char*)d_ws;
    float*  edge_card = (float*)(w + 0);          //  8192 f32
    float*  node_card = (float*)(w + 32768);      // 16384 f32
    float*  d0inv     = (float*)(w + 98304);      // 16384 f32
    float*  d1inv     = (float*)(w + 163840);     //  8192 f32
    bf16_t* W0t       = (bf16_t*)(w + 196608);    // 256x256 bf16 (c-major)
    bf16_t* W1t       = (bf16_t*)(w + 327680);    // 256x256 bf16
    bf16_t* Xb        = (bf16_t*)(w + 458752);    // 16384x256 bf16
    bf16_t* Yt        = (bf16_t*)(w + 8847360);   // 256x16384 bf16
    bf16_t* X1b       = (bf16_t*)(w + 17235968);  // 8192x256 bf16
    bf16_t* Zt        = (bf16_t*)(w + 21430272);  // 256x8192 bf16

    float* out0 = (float*)d_out;                  // relu(x_0_out) 16384x256
    float* out1 = out0 + (size_t)N_NODES * CH;    // relu(x_1)      8192x256

    k_zero<<<192, 256, 0, stream>>>(edge_card, 49152);
    k_convW<<<256, 256, 0, stream>>>(W0, W0t);
    k_convW<<<256, 256, 0, stream>>>(W1, W1t);
    k_convX<<<N_NODES, 256, 0, stream>>>(x0, Xb, N_NODES * CH);

    k_degrees<<<256, 256, 0, stream>>>(B1, edge_card, node_card);
    k_card<<<64, 256, 0, stream>>>(edge_card, node_card);
    k_dacc<<<256, 256, 0, stream>>>(B1, edge_card, node_card, d0inv, d1inv);
    k_invert<<<64, 256, 0, stream>>>(d0inv, d1inv);

    gemm_xw<<<N_NODES / 64, 256, 0, stream>>>(Xb, W0t, node_card, Yt, N_NODES);
    big_gemm<true><<<N_EDGES / 64, 256, 0, stream>>>(B1, Yt, d1inv, b01, out1, X1b);
    gemm_xw<<<N_EDGES / 64, 256, 0, stream>>>(X1b, W1t, edge_card, Zt, N_EDGES);
    big_gemm<false><<<N_NODES / 64, 256, 0, stream>>>(B1, Zt, d0inv, b10, out0, nullptr);
}
